// PlanetoidSAT_76879914598959
// MI455X (gfx1250) — matrix operand order repaired and checked by reading, not compiled, not run on hardware
//
#include <hip/hip_runtime.h>
#include <hip/hip_bf16.h>

typedef __attribute__((ext_vector_type(16))) _Float16 v16h;
typedef __attribute__((ext_vector_type(8)))  _Float16 v8h;
typedef __attribute__((ext_vector_type(8)))  float    v8f;

#define DEVINL __device__ __forceinline__

// monotone float<->uint encoding so atomicMax(uint) == float max
DEVINL unsigned fenc(float x) {
    unsigned u = __float_as_uint(x);
    return (u & 0x80000000u) ? ~u : (u | 0x80000000u);
}
DEVINL float fdec(unsigned u) {
    return (u & 0x80000000u) ? __uint_as_float(u & 0x7FFFFFFFu) : __uint_as_float(~u);
}

// ---------------------------------------------------------------- binarize
__global__ void k_binarize(const float* __restrict__ X, _Float16* __restrict__ Xh,
                           int n, int fin, int ldx) {
    long i = (long)blockIdx.x * blockDim.x + threadIdx.x;
    long total = (long)n * ldx;
    if (i >= total) return;
    int  col = (int)(i % ldx);
    long row = i / ldx;
    float v = (col < fin) ? X[row * fin + col] : 0.0f;
    Xh[i] = (_Float16)((v != 0.0f) ? 1.0f : 0.0f);
}

// ------------------------------------------------- pack W into B-fragment layout
// Wp[((kt*NTtot + ntBase + nt)*32 + lane)*16 + e]
// B 32x16 f16 layout: col=lane&15, h=lane>>4; K=8h+e (e<8), K=16+8h+(e-8) (e>=8)
__global__ void k_pack_w(const float* __restrict__ Wsrc, _Float16* __restrict__ Wp,
                         int ntBase, int NTtot, int KT, int Krows) {
    int tid = blockIdx.x * blockDim.x + threadIdx.x;
    if (tid >= KT * 2048) return;
    int e    = tid & 15;
    int lane = (tid >> 4) & 31;
    int nt   = (tid >> 9) & 3;
    int kt   = tid >> 11;
    int koff = (e < 8) ? (((lane >> 4) << 3) + e) : (16 + ((lane >> 4) << 3) + (e - 8));
    int k    = kt * 32 + koff;
    int col  = nt * 16 + (lane & 15);
    float v  = (k < Krows) ? Wsrc[(long)k * 64 + col] : 0.0f;
    Wp[(((long)kt * NTtot + ntBase + nt) * 32 + lane) * 16 + e] = (_Float16)v;
}

// ---------------------------------------------------------------- WMMA GEMM
// H[n, NT*16] = featA @ Wp + bias ; feat row built from 1..3 gathered X rows.
// MT = register blocking over M: each wave computes MT 16-row tiles so every
// B fragment load feeds MT WMMAs (vmem-per-wmma 1.25 instead of 2.25).
template <int NIDX, int NT, int MT>
__global__ __launch_bounds__(256)
void k_sat_gemm(const _Float16* __restrict__ Xh, int ldx,
                const int* __restrict__ idx,
                const _Float16* __restrict__ Wp,
                const float* __restrict__ bias,
                float* __restrict__ Hout,
                int n, int KT) {
    const int lane = threadIdx.x & 31;
    const int wid  = threadIdx.x >> 5;
    const int m0   = (blockIdx.x * 8 + wid) * (16 * MT);

    const _Float16* pA[MT][3];
#pragma unroll
    for (int t = 0; t < MT; ++t) {
        int row  = m0 + t * 16 + (lane & 15);
        int rowc = (row < n) ? row : (n - 1);
        int r0 = rowc, r1 = rowc, r2 = rowc;
        if (NIDX >= 2) { r0 = idx[NIDX * rowc]; r1 = idx[NIDX * rowc + 1]; }
        if (NIDX == 3) { r2 = idx[NIDX * rowc + 2]; }
        pA[t][0] = Xh + (long)r0 * ldx;
        pA[t][1] = Xh + (long)r1 * ldx;
        pA[t][2] = Xh + (long)r2 * ldx;
    }
    const int kshift = (lane < 16) ? 0 : 8;   // A 16x32 f16 ISA layout

    v8f acc[MT][NT] = {};

    for (int kt = 0; kt < KT; ++kt) {
        const int kb = kt * 32 + kshift;
        v16h a[MT];
#pragma unroll
        for (int t = 0; t < MT; ++t) {
            v8h lo = *(const v8h*)(pA[t][0] + kb);
            v8h hi = *(const v8h*)(pA[t][0] + kb + 16);
            if (NIDX >= 2) {
                lo *= *(const v8h*)(pA[t][1] + kb);
                hi *= *(const v8h*)(pA[t][1] + kb + 16);
            }
            if (NIDX == 3) {
                lo *= *(const v8h*)(pA[t][2] + kb);
                hi *= *(const v8h*)(pA[t][2] + kb + 16);
            }
#pragma unroll
            for (int i = 0; i < 8; ++i) { a[t][i] = lo[i]; a[t][8 + i] = hi[i]; }
        }

        const _Float16* wb = Wp + ((long)kt * NT * 32 + lane) * 16;
#pragma unroll
        for (int nt2 = 0; nt2 < NT; ++nt2) {
            v16h bfr = *(const v16h*)(wb + (long)nt2 * 32 * 16);
#pragma unroll
            for (int t = 0; t < MT; ++t) {
                acc[t][nt2] = __builtin_amdgcn_wmma_f32_16x16x32_f16(
                    false, a[t], false, bfr, (short)0, acc[t][nt2], false, false);
            }
        }
    }

    // C/D layout: vgpr j -> M = j (+8 for lanes 16..31), N = lane&15
    const int colBase = lane & 15;
    const int stride  = NT * 16;
#pragma unroll
    for (int t = 0; t < MT; ++t) {
        const int rbase = m0 + t * 16 + ((lane < 16) ? 0 : 8);
#pragma unroll
        for (int nt2 = 0; nt2 < NT; ++nt2) {
            const int col = nt2 * 16 + colBase;
            const float bv = bias[col];
#pragma unroll
            for (int j = 0; j < 8; ++j) {
                int r = rbase + j;
                if (r < n) Hout[(long)r * stride + col] = acc[t][nt2][j] + bv;
            }
        }
    }
}

// ---------------------------------------------------------------- scores
__global__ void k_scores(const float* __restrict__ H, int stride, int hoff,
                         const float* __restrict__ a1w, const float* __restrict__ a1b,
                         const float* __restrict__ a2w, const float* __restrict__ a2b,
                         float* __restrict__ s1, float* __restrict__ s2, int n) {
    int i = blockIdx.x * blockDim.x + threadIdx.x;
    if (i >= n) return;
    const float* h = H + (long)i * stride + hoff;
    float d1 = 0.f, d2 = 0.f;
#pragma unroll
    for (int k = 0; k < 64; ++k) {
        float v = h[k];
        d1 += v * a1w[k];
        d2 += v * a2w[k];
    }
    s1[i] = d1 + a1b[0];
    s2[i] = d2 + a2b[0];
}

// ---------------------------------------------------------------- attention passes
__global__ void k_att_max(const int* __restrict__ r, const int* __restrict__ c,
                          const float* __restrict__ s1, const float* __restrict__ s2,
                          unsigned* __restrict__ m, int nnz) {
    int j = blockIdx.x * blockDim.x + threadIdx.x;
    if (j >= nnz) return;
    int rr = r[j];
    atomicMax(&m[rr], fenc(s1[rr] + s2[c[j]]));
}

__global__ void k_att_exp(const int* __restrict__ r, const int* __restrict__ c,
                          const float* __restrict__ s1, const float* __restrict__ s2,
                          const unsigned* __restrict__ m, float* __restrict__ e,
                          float* __restrict__ z, int nnz) {
    int j = blockIdx.x * blockDim.x + threadIdx.x;
    if (j >= nnz) return;
    int rr = r[j];
    float ev = __expf(s1[rr] + s2[c[j]] - fdec(m[rr]));
    e[j] = ev;
    atomicAdd(&z[rr], ev);
}

__global__ void k_att_spmm(const int* __restrict__ r, const int* __restrict__ c,
                           const float* __restrict__ e, const float* __restrict__ z,
                           const float* __restrict__ H, int hstride, int hoff,
                           float* __restrict__ out, int nnz) {
    long t = (long)blockIdx.x * blockDim.x + threadIdx.x;
    long j = t >> 6;
    int  d = (int)(t & 63);
    if (j >= nnz) return;
    int rr = r[j], cc = c[j];
    float att = e[j] / z[rr];
    atomicAdd(&out[(long)rr * 64 + d], att * H[(long)cc * hstride + hoff + d]);
}

// ---------------------------------------------------------------- value SpMM
__global__ void k_spmm_val(const int* __restrict__ r, const int* __restrict__ c,
                           const float* __restrict__ val,
                           const float* __restrict__ X, int xstride,
                           float* __restrict__ out, int nnz) {
    long t = (long)blockIdx.x * blockDim.x + threadIdx.x;
    long j = t >> 6;
    int  d = (int)(t & 63);
    if (j >= nnz) return;
    atomicAdd(&out[(long)r[j] * 64 + d], val[j] * X[(long)c[j] * xstride + d]);
}

// ---------------------------------------------------------------- elementwise
__global__ void k_zero_f32(float* __restrict__ p, long count) {
    long i = (long)blockIdx.x * blockDim.x + threadIdx.x;
    if (i < count) p[i] = 0.0f;
}
__global__ void k_copy_f32(float* __restrict__ d, const float* __restrict__ s, int count) {
    int i = blockIdx.x * blockDim.x + threadIdx.x;
    if (i < count) d[i] = s[i];
}
__global__ void k_scale_f32(float* __restrict__ p, float s, long count) {
    long i = (long)blockIdx.x * blockDim.x + threadIdx.x;
    if (i < count) p[i] *= s;
}
__global__ void k_f32_to_f16(const float* __restrict__ s, _Float16* __restrict__ d, long count) {
    long i = (long)blockIdx.x * blockDim.x + threadIdx.x;
    if (i < count) d[i] = (_Float16)s[i];
}
__global__ void k_prelu_add(const float* __restrict__ A, const float* __restrict__ B,
                            const float* __restrict__ pw, float* __restrict__ O, long count) {
    long i = (long)blockIdx.x * blockDim.x + threadIdx.x;
    if (i >= count) return;
    float x = A[i] + B[i];
    O[i] = (x >= 0.0f) ? x : pw[0] * x;
}

// ================================================================ launcher
extern "C" void kernel_launch(void* const* d_in, const int* in_sizes, int n_in,
                              void* d_out, int out_size, void* d_ws, size_t ws_size,
                              hipStream_t stream) {
    (void)in_sizes; (void)n_in; (void)out_size; (void)ws_size;
    const int N0 = 50000, N1 = 200000, N2 = 100000, FIN = 500, D = 64;
    const int NNZ0 = 600000, NNZ1 = 1000000, NNZ2 = 600000;
    const int KT = 16, LDX = 512, TB = 256;
    const int ROWS_PER_BLOCK = 8 * 16 * 2;   // 8 waves * 16 rows * MT(2)

    const float* X0   = (const float*)d_in[0];
    const int*   E1   = (const int*)d_in[1];
    const int*   T2   = (const int*)d_in[2];
    const int*   L0   = (const int*)d_in[3];
    const int*   L1a  = (const int*)d_in[4];
    const int*   L1b  = (const int*)d_in[5];
    const int*   L2   = (const int*)d_in[6];
    const int*   B1i  = (const int*)d_in[7];
    const float* B1v  = (const float*)d_in[8];
    const int*   B2i  = (const int*)d_in[9];
    const float* B2v  = (const float*)d_in[10];
    const float* W    = (const float*)d_in[11];
    const float* bb   = (const float*)d_in[12];
    const float* a1w  = (const float*)d_in[13];
    const float* a1b  = (const float*)d_in[14];
    const float* a2w  = (const float*)d_in[15];
    const float* a2b  = (const float*)d_in[16];
    const float* pw   = (const float*)d_in[17];
    const float* triW = (const float*)d_in[18];
    const float* trib = (const float*)d_in[19];

    char* base = (char*)d_ws;
    size_t off = 0;
    auto alloc = [&](size_t bytes) -> char* {
        char* p = base + off;
        off = (off + bytes + 255) & ~(size_t)255;
        return p;
    };

    _Float16* Xh    = (_Float16*)alloc((size_t)N0 * LDX * 2);
    _Float16* Wp    = (_Float16*)alloc((size_t)KT * 8 * 32 * 16 * 2);
    float*    biasP = (float*)alloc(128 * 4);
    _Float16* TriWp = (_Float16*)alloc((size_t)2 * 4 * 32 * 16 * 2);
    float*    hPair = (float*)alloc((size_t)N1 * 128 * 4);
    float*    outA  = (float*)alloc((size_t)N1 * 64 * 4);
    float*    outB  = (float*)alloc((size_t)N1 * 64 * 4);
    float*    s1    = (float*)alloc((size_t)N1 * 4);
    float*    s2    = (float*)alloc((size_t)N1 * 4);
    unsigned* mbuf  = (unsigned*)alloc((size_t)N1 * 4);
    float*    zbuf  = (float*)alloc((size_t)N1 * 4);
    float*    ebuf  = (float*)alloc((size_t)NNZ1 * 4);
    float*    H1    = (float*)alloc((size_t)N1 * 64 * 4);
    float*    H2    = (float*)alloc((size_t)N2 * 64 * 4);
    float*    Hout  = (float*)d_out;
    // region reuse after the three SAT pairs are done:
    float*    BH2   = hPair;                                   // [N1*64] f32
    _Float16* BH2h  = (_Float16*)(hPair + (size_t)N1 * 64);    // [N1*64] f16
    float*    Tm    = outA;                                    // [N1*64] f32

    auto cdiv = [](long a, long b) { return (int)((a + b - 1) / b); };

    // 1. binarize node features, pad K 500 -> 512
    k_binarize<<<cdiv((long)N0 * LDX, TB), TB, 0, stream>>>(X0, Xh, N0, FIN, LDX);

    auto run_head_att = [&](int head, int hoff, const int* idx, int nnz, int n, float* out) {
        k_scores<<<cdiv(n, TB), TB, 0, stream>>>(hPair, 128, hoff, a1w + head * 64, a1b + head,
                                                 a2w + head * 64, a2b + head, s1, s2, n);
        k_zero_f32<<<cdiv(n, TB), TB, 0, stream>>>((float*)mbuf, n);
        k_zero_f32<<<cdiv(n, TB), TB, 0, stream>>>(zbuf, n);
        k_zero_f32<<<cdiv((long)n * 64, TB), TB, 0, stream>>>(out, (long)n * 64);
        const int* r = idx;
        const int* c = idx + nnz;
        k_att_max<<<cdiv(nnz, TB), TB, 0, stream>>>(r, c, s1, s2, mbuf, nnz);
        k_att_exp<<<cdiv(nnz, TB), TB, 0, stream>>>(r, c, s1, s2, mbuf, ebuf, zbuf, nnz);
        k_att_spmm<<<cdiv((long)nnz * 64, TB), TB, 0, stream>>>(r, c, ebuf, zbuf, hPair, 128,
                                                                hoff, out, nnz);
    };

    auto run_pair = [&](int hA, int hB, int NIDX, const int* featIdx, int n,
                        const int* idxA, int nnzA, const int* idxB, int nnzB, float* Hdst) {
        k_pack_w<<<cdiv(KT * 2048, TB), TB, 0, stream>>>(W + (size_t)hA * FIN * D, Wp, 0, 8, KT, FIN);
        k_pack_w<<<cdiv(KT * 2048, TB), TB, 0, stream>>>(W + (size_t)hB * FIN * D, Wp, 4, 8, KT, FIN);
        k_copy_f32<<<1, 64, 0, stream>>>(biasP, bb + hA * D, 64);
        k_copy_f32<<<1, 64, 0, stream>>>(biasP + 64, bb + hB * D, 64);
        int grid = cdiv(n, ROWS_PER_BLOCK);
        if (NIDX == 1)
            k_sat_gemm<1, 8, 2><<<grid, TB, 0, stream>>>(Xh, LDX, nullptr, Wp, biasP, hPair, n, KT);
        else if (NIDX == 2)
            k_sat_gemm<2, 8, 2><<<grid, TB, 0, stream>>>(Xh, LDX, featIdx, Wp, biasP, hPair, n, KT);
        else
            k_sat_gemm<3, 8, 2><<<grid, TB, 0, stream>>>(Xh, LDX, featIdx, Wp, biasP, hPair, n, KT);
        run_head_att(hA, 0, idxA, nnzA, n, outA);
        run_head_att(hB, 64, idxB, nnzB, n, outB);
        k_prelu_add<<<cdiv((long)n * 64, TB), TB, 0, stream>>>(outA, outB, pw, Hdst, (long)n * 64);
    };

    // H0 -> d_out directly; H1, H2 -> workspace
    run_pair(0, 1, 1, nullptr, N0, L0, NNZ0, L0, NNZ0, Hout);
    run_pair(2, 3, 2, E1, N1, L1a, NNZ1, L1b, NNZ1, H1);
    run_pair(4, 5, 3, T2, N2, L2, NNZ2, L2, NNZ2, H2);

    // Tm = (B2 @ H2) @ triW + trib
    k_zero_f32<<<cdiv((long)N1 * 64, TB), TB, 0, stream>>>(BH2, (long)N1 * 64);
    k_spmm_val<<<cdiv((long)3 * N2 * 64, TB), TB, 0, stream>>>(B2i, B2i + 3 * N2, B2v, H2, 64,
                                                               BH2, 3 * N2);
    k_f32_to_f16<<<cdiv((long)N1 * 64, TB), TB, 0, stream>>>(BH2, BH2h, (long)N1 * 64);
    k_pack_w<<<cdiv(2 * 2048, TB), TB, 0, stream>>>(triW, TriWp, 0, 4, 2, 64);
    k_copy_f32<<<1, 64, 0, stream>>>(biasP, trib, 64);
    k_sat_gemm<1, 4, 2><<<cdiv(N1, ROWS_PER_BLOCK), TB, 0, stream>>>(BH2h, 64, nullptr, TriWp,
                                                                     biasP, Tm, N1, 2);

    // out = (H0 + B1@H1 + B1@Tm) / 3 ; d_out already holds H0
    k_spmm_val<<<cdiv((long)2 * N1 * 64, TB), TB, 0, stream>>>(B1i, B1i + 2 * N1, B1v, H1, 64,
                                                               Hout, 2 * N1);
    k_spmm_val<<<cdiv((long)2 * N1 * 64, TB), TB, 0, stream>>>(B1i, B1i + 2 * N1, B1v, Tm, 64,
                                                               Hout, 2 * N1);
    k_scale_f32<<<cdiv((long)N0 * 64, TB), TB, 0, stream>>>(Hout, 1.0f / 3.0f, (long)N0 * 64);
}
